// ParallelSelfAttention_85581518340497
// MI455X (gfx1250) — matrix operand order repaired and checked by reading, not compiled, not run on hardware
//
#include <hip/hip_runtime.h>

// ---------------------------------------------------------------------------
// Types for CDNA5 WMMA (wave32, v_wmma_f32_16x16x32_bf16)
// ---------------------------------------------------------------------------
typedef __bf16 bf16;
typedef __attribute__((ext_vector_type(16))) __bf16 v16bf;
typedef __attribute__((ext_vector_type(8)))  __bf16 bf16x8;
typedef __attribute__((ext_vector_type(8)))  float  v8f;

#define WMMA_BF16(A, B, C) \
  __builtin_amdgcn_wmma_f32_16x16x32_bf16(false, (A), false, (B), (short)0, (C), false, false)

union AFrag {
  v16bf v;
  struct { bf16x8 lo, hi; } p;
};

static __device__ __forceinline__ bf16 f2bf(float f) {
  union { float f; unsigned int u; } in;
  in.f = f;
  unsigned int u = in.u + 0x7FFFu + ((in.u >> 16) & 1u);   // round-to-nearest-even
  union { unsigned short s; bf16 b; } out;
  out.s = (unsigned short)(u >> 16);
  return out.b;
}

// Problem constants
#define SEQ   2048
#define HID   2048
#define NH    16
#define HD    128
#define M_TOT 4096          /* B*S */
#define N_QKV 6144          /* 3*HID */

// ---------------------------------------------------------------------------
// Kernel 1: cast hidden_states fp32 -> bf16 (row-major [M_TOT][HID])
// ---------------------------------------------------------------------------
__global__ __launch_bounds__(256)
void pack_x_kernel(const float* __restrict__ in, bf16* __restrict__ out, int n) {
  int i = blockIdx.x * 256 + threadIdx.x;
  if (i < n) out[i] = f2bf(in[i]);
}

// ---------------------------------------------------------------------------
// Kernel 2/3: transpose-cast weight fp32 [K][N] -> bf16 [N][K] (N-major,
// so WMMA B-fragments are two contiguous 16B k-runs per lane)
// ---------------------------------------------------------------------------
__global__ __launch_bounds__(256)
void pack_wt_kernel(const float* __restrict__ in, bf16* __restrict__ out, int K, int N) {
  size_t i = (size_t)blockIdx.x * 256 + threadIdx.x;
  if (i >= (size_t)N * K) return;
  int n = (int)(i / K);
  int k = (int)(i % K);
  out[i] = f2bf(in[(size_t)k * N + n]);
}

// ---------------------------------------------------------------------------
// Kernel 4: QKV GEMM.  Xb [M_TOT][HID] bf16  x  WqkvT [N_QKV][HID] bf16
// 4 waves / block; wave tile 32x64 -> block tile 64x128.
// Epilogue scatters: Q,K row-major per head; V transposed [bh][d][s].
// ---------------------------------------------------------------------------
__global__ __launch_bounds__(128)
void gemm_qkv_kernel(const bf16* __restrict__ Xb, const bf16* __restrict__ Wt,
                     const float* __restrict__ bias,
                     bf16* __restrict__ Qp, bf16* __restrict__ Kp, bf16* __restrict__ Vt) {
  const int lane = threadIdx.x & 31;
  const int w    = threadIdx.x >> 5;
  const int half = lane >> 4;
  const int lr   = lane & 15;
  const int m0 = blockIdx.y * 64  + (w >> 1) * 32;
  const int n0 = blockIdx.x * 128 + (w & 1) * 64;

  v8f acc[2][4] = {};

  for (int kb = 0; kb < HID; kb += 32) {
    AFrag a[2];
#pragma unroll
    for (int mi = 0; mi < 2; ++mi) {
      const bf16* ar = Xb + (size_t)(m0 + mi * 16 + lr) * HID + kb;
      a[mi].p.lo = *(const bf16x8*)(ar + half * 8);
      a[mi].p.hi = *(const bf16x8*)(ar + 16 + half * 8);
      __builtin_prefetch(ar + 32, 0, 1);
    }
    v16bf bfrag[4];
#pragma unroll
    for (int t = 0; t < 4; ++t) {
      const bf16* br = Wt + (size_t)(n0 + t * 16 + lr) * HID + kb + half * 8;
      bfrag[t] = __builtin_shufflevector(*(const bf16x8*)br, *(const bf16x8*)(br + 16), 0, 1, 2, 3, 4, 5, 6, 7, 8, 9, 10, 11, 12, 13, 14, 15);
      __builtin_prefetch(br + 32, 0, 1);
    }
#pragma unroll
    for (int mi = 0; mi < 2; ++mi)
#pragma unroll
      for (int t = 0; t < 4; ++t)
        acc[mi][t] = WMMA_BF16(a[mi].v, bfrag[t], acc[mi][t]);
  }

#pragma unroll
  for (int mi = 0; mi < 2; ++mi)
    for (int t = 0; t < 4; ++t)
#pragma unroll
      for (int r = 0; r < 8; ++r) {
        int m = m0 + mi * 16 + r + half * 8;
        int n = n0 + t * 16 + lr;
        float val = acc[mi][t][r] + bias[n];
        int sec = n >> 11, nn = n & 2047, hd = nn >> 7, dd = nn & 127;
        int b = m >> 11, s = m & 2047;
        bf16 o = f2bf(val);
        size_t bh = (size_t)b * NH + hd;
        if (sec == 0)      Qp[(bh * SEQ + s) * HD + dd] = o;
        else if (sec == 1) Kp[(bh * SEQ + s) * HD + dd] = o;
        else               Vt[(bh * HD + dd) * SEQ + s] = o;
      }
}

// ---------------------------------------------------------------------------
// Kernel 5: flash attention.  1024 blocks = (b,h) x 32 q-tiles-of-64.
// 4 waves/block, each wave owns a 16-query tile, streams 32 keys per step.
// Online softmax with wave32 half-shuffles; P staged via wave-private LDS
// to convert C-layout -> A-layout for the P@V WMMAs.
// ---------------------------------------------------------------------------
__global__ __launch_bounds__(128)
void attn_kernel(const bf16* __restrict__ Qp, const bf16* __restrict__ Kp,
                 const bf16* __restrict__ Vt, const float* __restrict__ maskp,
                 bf16* __restrict__ Ctx) {
  __shared__ __align__(64) bf16 Plds[4 * 16 * 32];   // 4KB: 16x32 P tile per wave

  const int lane = threadIdx.x & 31;
  const int w    = threadIdx.x >> 5;
  const int half = lane >> 4;
  const int lr   = lane & 15;

  const int bh = blockIdx.x >> 5;       // b*16 + h
  const int qt = blockIdx.x & 31;
  const int b  = bh >> 4;
  const int h  = bh & 15;
  const int s0 = qt * 64 + w * 16;
  const float scale = 0.08838834764831845f;  // 1/sqrt(128)

  // Q fragments (16 x 128 = 4 A-frags), kept in registers
  AFrag qf[4];
  const bf16* qbase = Qp + ((size_t)bh * SEQ + s0 + lr) * HD;
#pragma unroll
  for (int t = 0; t < 4; ++t) {
    qf[t].p.lo = *(const bf16x8*)(qbase + t * 32 + half * 8);
    qf[t].p.hi = *(const bf16x8*)(qbase + t * 32 + 16 + half * 8);
  }

  float mrow[8], lrow[8];
#pragma unroll
  for (int r = 0; r < 8; ++r) { mrow[r] = -3.0e38f; lrow[r] = 0.0f; }
  v8f acc[8] = {};                       // ctx accumulator: 16 x 128

  bf16* myP = &Plds[w * 512];
  const float* mbase = maskp + (size_t)b * SEQ;

  for (int j = 0; j < SEQ; j += 32) {
    // ---- scores: two 16x16 tiles (keys j..j+15, j+16..j+31) ----
    v8f sc[2] = {};
#pragma unroll
    for (int nt = 0; nt < 2; ++nt) {
      const bf16* kb = Kp + ((size_t)bh * SEQ + j + nt * 16 + lr) * HD + half * 8;
#pragma unroll
      for (int t = 0; t < 4; ++t) {
        v16bf kf = __builtin_shufflevector(*(const bf16x8*)(kb + t * 32), *(const bf16x8*)(kb + t * 32 + 16), 0, 1, 2, 3, 4, 5, 6, 7, 8, 9, 10, 11, 12, 13, 14, 15);
        sc[nt] = WMMA_BF16(qf[t].v, kf, sc[nt]);
      }
    }
    float mk0 = mbase[j + lr];
    float mk1 = mbase[j + 16 + lr];

    float x0[8], x1[8];
#pragma unroll
    for (int r = 0; r < 8; ++r) {
      x0[r] = sc[0][r] * scale + mk0;
      x1[r] = sc[1][r] * scale + mk1;
    }

    // ---- online softmax (reduce across the 16 lanes of each half) ----
    float alpha[8];
#pragma unroll
    for (int r = 0; r < 8; ++r) {
      float t = fmaxf(x0[r], x1[r]);
      for (int m = 8; m >= 1; m >>= 1) t = fmaxf(t, __shfl_xor(t, m, 32));
      float newm = fmaxf(mrow[r], t);
      alpha[r] = __expf(mrow[r] - newm);
      float p0 = __expf(x0[r] - newm);
      float p1 = __expf(x1[r] - newm);
      float s  = p0 + p1;
      for (int m = 8; m >= 1; m >>= 1) s += __shfl_xor(s, m, 32);
      lrow[r] = lrow[r] * alpha[r] + s;
      mrow[r] = newm;
      int row = r + half * 8;
      myP[row * 32 + lr]      = f2bf(p0);   // C-layout -> LDS (row-major 16x32)
      myP[row * 32 + 16 + lr] = f2bf(p1);
    }
#pragma unroll
    for (int nj = 0; nj < 8; ++nj)
#pragma unroll
      for (int r = 0; r < 8; ++r) acc[nj][r] *= alpha[r];

    // ---- reload P in A-fragment layout (wave-private, LDS in-order) ----
    AFrag pf;
    pf.p.lo = *(const bf16x8*)&myP[lr * 32 + half * 8];
    pf.p.hi = *(const bf16x8*)&myP[lr * 32 + 16 + half * 8];

    // ---- ctx += P @ V  (V stored d-major so B-frags are contiguous) ----
#pragma unroll
    for (int nj = 0; nj < 8; ++nj) {
      const bf16* vp = Vt + ((size_t)bh * HD + nj * 16 + lr) * SEQ + j + half * 8;
      v16bf vf = __builtin_shufflevector(*(const bf16x8*)vp, *(const bf16x8*)(vp + 16), 0, 1, 2, 3, 4, 5, 6, 7, 8, 9, 10, 11, 12, 13, 14, 15);
      acc[nj] = WMMA_BF16(pf.v, vf, acc[nj]);
    }
  }

  // ---- normalize and store ctx as bf16 [B*S][HID] ----
  float inv[8];
#pragma unroll
  for (int r = 0; r < 8; ++r) inv[r] = 1.0f / lrow[r];
#pragma unroll
  for (int nj = 0; nj < 8; ++nj)
#pragma unroll
    for (int r = 0; r < 8; ++r) {
      int row = s0 + r + half * 8;
      int col = h * HD + nj * 16 + lr;
      Ctx[((size_t)b * SEQ + row) * HID + col] = f2bf(acc[nj][r] * inv[r]);
    }
}

// ---------------------------------------------------------------------------
// Kernel 6: output projection.  Ctx [M_TOT][HID] bf16 x WoutT [HID][HID] bf16
// -> out fp32 + bias.
// ---------------------------------------------------------------------------
__global__ __launch_bounds__(128)
void gemm_out_kernel(const bf16* __restrict__ Cb, const bf16* __restrict__ Wt,
                     const float* __restrict__ bias, float* __restrict__ out) {
  const int lane = threadIdx.x & 31;
  const int w    = threadIdx.x >> 5;
  const int half = lane >> 4;
  const int lr   = lane & 15;
  const int m0 = blockIdx.y * 64  + (w >> 1) * 32;
  const int n0 = blockIdx.x * 128 + (w & 1) * 64;

  v8f acc[2][4] = {};
  for (int kb = 0; kb < HID; kb += 32) {
    AFrag a[2];
#pragma unroll
    for (int mi = 0; mi < 2; ++mi) {
      const bf16* ar = Cb + (size_t)(m0 + mi * 16 + lr) * HID + kb;
      a[mi].p.lo = *(const bf16x8*)(ar + half * 8);
      a[mi].p.hi = *(const bf16x8*)(ar + 16 + half * 8);
    }
    v16bf bfrag[4];
#pragma unroll
    for (int t = 0; t < 4; ++t)
      bfrag[t] = __builtin_shufflevector(*(const bf16x8*)(Wt + (size_t)(n0 + t * 16 + lr) * HID + kb + half * 8), *(const bf16x8*)(Wt + (size_t)(n0 + t * 16 + lr) * HID + kb + 16 + half * 8), 0, 1, 2, 3, 4, 5, 6, 7, 8, 9, 10, 11, 12, 13, 14, 15);
#pragma unroll
    for (int mi = 0; mi < 2; ++mi)
#pragma unroll
      for (int t = 0; t < 4; ++t)
        acc[mi][t] = WMMA_BF16(a[mi].v, bfrag[t], acc[mi][t]);
  }

#pragma unroll
  for (int mi = 0; mi < 2; ++mi)
    for (int t = 0; t < 4; ++t)
#pragma unroll
      for (int r = 0; r < 8; ++r) {
        int m = m0 + mi * 16 + r + half * 8;
        int n = n0 + t * 16 + lr;
        out[(size_t)m * HID + n] = acc[mi][t][r] + bias[n];
      }
}

// ---------------------------------------------------------------------------
// Host launch
// ---------------------------------------------------------------------------
extern "C" void kernel_launch(void* const* d_in, const int* in_sizes, int n_in,
                              void* d_out, int out_size, void* d_ws, size_t ws_size,
                              hipStream_t stream) {
  (void)in_sizes; (void)n_in; (void)out_size; (void)ws_size;
  const float* hidden = (const float*)d_in[0];
  const float* mask   = (const float*)d_in[1];
  const float* W_qkv  = (const float*)d_in[2];
  const float* b_qkv  = (const float*)d_in[3];
  const float* W_out  = (const float*)d_in[4];
  const float* b_out  = (const float*)d_in[5];
  float* out = (float*)d_out;

  char* ws = (char*)d_ws;
  bf16* Xb    = (bf16*)(ws);                     // 16 MB  [4096][2048]
  bf16* WqkvT = (bf16*)(ws + (16u << 20));       // 24 MB  [6144][2048]
  bf16* WoutT = (bf16*)(ws + (40u << 20));       //  8 MB  [2048][2048]
  bf16* Qp    = (bf16*)(ws + (48u << 20));       // 16 MB  [32][2048][128]
  bf16* Kp    = (bf16*)(ws + (64u << 20));       // 16 MB  [32][2048][128]
  bf16* Vt    = (bf16*)(ws + (80u << 20));       // 16 MB  [32][128][2048]
  bf16* Ctx   = (bf16*)(ws + (96u << 20));       // 16 MB  [4096][2048]

  pack_x_kernel<<<(M_TOT * HID) / 256, 256, 0, stream>>>(hidden, Xb, M_TOT * HID);
  pack_wt_kernel<<<(HID * N_QKV) / 256, 256, 0, stream>>>(W_qkv, WqkvT, HID, N_QKV);
  pack_wt_kernel<<<(HID * HID) / 256, 256, 0, stream>>>(W_out, WoutT, HID, HID);

  gemm_qkv_kernel<<<dim3(N_QKV / 128, M_TOT / 64), 128, 0, stream>>>(
      Xb, WqkvT, b_qkv, Qp, Kp, Vt);

  attn_kernel<<<2 * NH * (SEQ / 64), 128, 0, stream>>>(Qp, Kp, Vt, mask, Ctx);

  gemm_out_kernel<<<dim3(HID / 128, M_TOT / 64), 128, 0, stream>>>(
      Ctx, WoutT, b_out, out);
}
